// MultiHeadAttention_12738873000630
// MI455X (gfx1250) — compile-verified, matrix operand order checked by reading, not run on hardware
//
#include <hip/hip_runtime.h>
#include <stdint.h>

#define TSEQ   2048
#define CDIM   1024
#define NHEAD  16
#define DHEAD  64
#define BATCH  4
#define MROWS  (BATCH * TSEQ)
#define SCL    0.125f          /* 1/sqrt(64) */
#define NEGINF (-1e9f)

typedef __attribute__((ext_vector_type(16))) __bf16 v16bf;
typedef __attribute__((ext_vector_type(8)))  __bf16 v8bf;
typedef __attribute__((ext_vector_type(4)))  __bf16 v4bf;
typedef __attribute__((ext_vector_type(8)))  float  v8f;

union V16U { v16bf v; v8bf h[2]; };

__device__ __forceinline__ v8f wmma_bf16(v16bf a, v16bf b, v8f c) {
  return __builtin_amdgcn_wmma_f32_16x16x32_bf16(false, a, false, b,
                                                 (short)0, c, false, false);
}

// Load a 16x32 bf16 tile from row-major memory into the WMMA A-operand
// register layout (also the B-operand pattern when the source holds the
// transposed B matrix row-major).
__device__ __forceinline__ v16bf load_a16x32(const __bf16* __restrict__ src,
                                             int ld, int row0, int col0) {
  const int lane = threadIdx.x & 31;
  const __bf16* p = src + (size_t)(row0 + (lane & 15)) * (size_t)ld
                        + (size_t)(col0 + ((lane >> 4) << 3));
  V16U u;
  u.h[0] = *(const v8bf*)(p);        // K = cbase .. cbase+7
  u.h[1] = *(const v8bf*)(p + 16);   // K = cbase+16 .. cbase+23
  return u.v;
}

// Same A-operand pattern, reading a row-major 16x32 tile resident in LDS.
__device__ __forceinline__ v16bf load_a_lds(const __bf16* p) {
  V16U u;
  u.h[0] = *(const v8bf*)(p);
  u.h[1] = *(const v8bf*)(p + 16);
  return u.v;
}

// Async copy of 512B (32 lanes x 16B) global -> LDS, tracked by ASYNCcnt.
__device__ __forceinline__ void async_b128(unsigned lds_addr, uint64_t gaddr) {
  asm volatile("global_load_async_to_lds_b128 %0, %1, off"
               :: "v"(lds_addr), "v"(gaddr) : "memory");
}

__device__ __forceinline__ float redmax16(float v) {
  v = fmaxf(v, __shfl_xor(v, 1, 32));
  v = fmaxf(v, __shfl_xor(v, 2, 32));
  v = fmaxf(v, __shfl_xor(v, 4, 32));
  v = fmaxf(v, __shfl_xor(v, 8, 32));
  return v;
}
__device__ __forceinline__ float redsum16(float v) {
  v += __shfl_xor(v, 1, 32);
  v += __shfl_xor(v, 2, 32);
  v += __shfl_xor(v, 4, 32);
  v += __shfl_xor(v, 8, 32);
  return v;
}

// ---------------------------------------------------------------- converters
__global__ __launch_bounds__(256)
void cvt_f32_bf16(const float* __restrict__ x, __bf16* __restrict__ xb, int n4) {
  int i = blockIdx.x * 256 + threadIdx.x;
  if (i >= n4) return;
  const float4 f = ((const float4*)x)[i];
  v4bf pv;
  pv[0] = (__bf16)f.x; pv[1] = (__bf16)f.y;
  pv[2] = (__bf16)f.z; pv[3] = (__bf16)f.w;
  ((v4bf*)xb)[i] = pv;
}

// W [C,C] f32 row-major  ->  Wt [C,C] bf16 with Wt[n][k] = W[k][n]
__global__ __launch_bounds__(256)
void w_transpose_bf16(const float* __restrict__ W, __bf16* __restrict__ Wt) {
  __shared__ float tile[32][33];
  const int tx = threadIdx.x & 31, ty = threadIdx.x >> 5;
  const int bx = blockIdx.x & 31;   // n tile
  const int by = blockIdx.x >> 5;   // k tile
#pragma unroll
  for (int i = 0; i < 4; ++i)
    tile[ty + 8 * i][tx] = W[(size_t)(by * 32 + ty + 8 * i) * CDIM + bx * 32 + tx];
  __syncthreads();
#pragma unroll
  for (int i = 0; i < 4; ++i)
    Wt[(size_t)(bx * 32 + ty + 8 * i) * CDIM + by * 32 + tx] =
        (__bf16)tile[tx][ty + 8 * i];
}

// ------------------------------------------------------------- WMMA GEMM
// Y[16x64 tile] = X[m0:m0+16, :] * Wt^T + bias.  One wave per tile; the 4
// waves of a block share m0, so the A tile is staged once in LDS via
// double-buffered async copies (ASYNCcnt) issued by wave 0.
// mode 0: bf16 out, layout [B,H,T,D]      (q / k)
// mode 1: bf16 out, layout [B,H,D,T]      (v transposed)
// mode 2: f32  out, layout [M, C]         (final projection)
__global__ __launch_bounds__(128)
void gemm_wmma(const __bf16* __restrict__ X, const __bf16* __restrict__ Wt,
               const float* __restrict__ bias, void* __restrict__ dst, int mode) {
  __shared__ __bf16 shA[2][16 * 32];          // 2KB double-buffered A tile
  const int lane = threadIdx.x & 31;
  const int wv   = threadIdx.x >> 5;
  const int wid  = blockIdx.x * 4 + wv;       // 0..8191
  const int m0   = (wid >> 4) * 16;           // shared by the 4 waves
  const int nb   = wid & 15;                  // 16 n blocks (== head id)
  const int n0   = nb * 64;
  const int half = lane >> 4, l15 = lane & 15;

  // linear-copy addressing for the 16x32 bf16 (1KB) A tile: two 512B passes
  const uint64_t xrow0 = (uint64_t)(X + (size_t)(m0 + (lane >> 2)) * CDIM
                                      + (lane & 3) * 8);
  const uint64_t xrow1 = (uint64_t)(X + (size_t)(m0 + 8 + (lane >> 2)) * CDIM
                                      + (lane & 3) * 8);
  const unsigned ldsA  = (unsigned)(uintptr_t)(&shA[0][0]) + lane * 16;

  if (wv == 0) {                               // prologue: stage k0 = 0
    async_b128(ldsA,       xrow0);
    async_b128(ldsA + 512, xrow1);
  }

  v8f acc[4] = {};
  int buf = 0;
  for (int k0 = 0; k0 < CDIM; k0 += 32) {
    if (wv == 0) {
      if (k0 + 32 < CDIM) {                    // stage next tile into buf^1
        const unsigned l = ldsA + (buf ^ 1) * 1024;
        async_b128(l,       xrow0 + (k0 + 32) * 2);
        async_b128(l + 512, xrow1 + (k0 + 32) * 2);
        asm volatile("s_wait_asynccnt 0x2" ::: "memory");  // current tile done
      } else {
        asm volatile("s_wait_asynccnt 0x0" ::: "memory");
      }
    }
    __syncthreads();                           // A tile visible to all waves
    const v16bf A = load_a_lds(&shA[buf][l15 * 32 + (half << 3)]);
#pragma unroll
    for (int j = 0; j < 4; ++j) {
      const v16bf Bt = load_a16x32(Wt, CDIM, n0 + 16 * j, k0);
      acc[j] = wmma_bf16(A, Bt, acc[j]);
    }
    __syncthreads();                           // reads done before overwrite
    buf ^= 1;
  }

  float bj[4];
#pragma unroll
  for (int j = 0; j < 4; ++j) bj[j] = bias[n0 + 16 * j + l15];

  if (mode == 0) {
    __bf16* o = (__bf16*)dst;
#pragma unroll
    for (int r = 0; r < 8; ++r) {
      const int m = m0 + r + 8 * half;
      const int b = m >> 11, t = m & (TSEQ - 1);
      const size_t base = (((size_t)b * NHEAD + nb) * TSEQ + t) * DHEAD;
#pragma unroll
      for (int j = 0; j < 4; ++j)
        o[base + 16 * j + l15] = (__bf16)(acc[j][r] + bj[j]);
    }
  } else if (mode == 1) {
    __bf16* o = (__bf16*)dst;
    const int b  = m0 >> 11;
    const int tb = (m0 & (TSEQ - 1)) + 8 * half;
#pragma unroll
    for (int j = 0; j < 4; ++j) {
      const int d = 16 * j + l15;
      v8bf pk;
#pragma unroll
      for (int r = 0; r < 8; ++r) pk[r] = (__bf16)(acc[j][r] + bj[j]);
      *(v8bf*)&o[(((size_t)b * NHEAD + nb) * DHEAD + d) * TSEQ + tb] = pk;
    }
  } else {
    float* o = (float*)dst;
#pragma unroll
    for (int r = 0; r < 8; ++r) {
      const int m = m0 + r + 8 * half;
#pragma unroll
      for (int j = 0; j < 4; ++j)
        o[(size_t)m * CDIM + n0 + 16 * j + l15] = acc[j][r] + bj[j];
    }
  }
}

// --------------------------------------------------------- flash attention
// One wave per (b, h, 16-query tile).  Causal loop over 32-key blocks.
__global__ __launch_bounds__(128)
void attn_wmma(const __bf16* __restrict__ qb, const __bf16* __restrict__ kb,
               const __bf16* __restrict__ vtb, const int* __restrict__ amask,
               __bf16* __restrict__ yb) {
  __shared__ __bf16 psc[4][16 * 32];          // per-wave P transpose scratch
  const int lane = threadIdx.x & 31, wv = threadIdx.x >> 5;
  const int wid  = blockIdx.x * 4 + wv;       // 0..8191
  const int qt   = wid & 127;
  const int h    = (wid >> 7) & (NHEAD - 1);
  const int b    = wid >> 11;
  const int t0   = qt * 16;
  const int half = lane >> 4, l15 = lane & 15;

  const size_t ho = (size_t)b * NHEAD + h;
  const __bf16* qh = qb  + ho * (size_t)(TSEQ * DHEAD);
  const __bf16* kh = kb  + ho * (size_t)(TSEQ * DHEAD);
  const __bf16* vh = vtb + ho * (size_t)(DHEAD * TSEQ);   // [D, T]

  const v16bf qA0 = load_a16x32(qh, DHEAD, t0, 0);
  const v16bf qA1 = load_a16x32(qh, DHEAD, t0, 32);

  v8f O[4] = {};
  float rmax[8], rsum[8];
#pragma unroll
  for (int r = 0; r < 8; ++r) { rmax[r] = -3.0e38f; rsum[r] = 0.f; }

  const int thi = t0 + 15;
  for (int k0 = 0; k0 <= thi; k0 += 32) {
    if (k0 + 32 <= thi) {                     // hide latency of next block
      __builtin_prefetch(kh + (size_t)(k0 + 32) * DHEAD, 0, 3);
      __builtin_prefetch(vh + (size_t)(k0 + 32), 0, 3);
    }
    v8f z = {};
    v16bf kB = load_a16x32(kh, DHEAD, k0, 0);
    v8f S0 = wmma_bf16(qA0, kB, z);
    kB = load_a16x32(kh, DHEAD, k0, 32);
    S0 = wmma_bf16(qA1, kB, S0);
    kB = load_a16x32(kh, DHEAD, k0 + 16, 0);
    v8f S1 = wmma_bf16(qA0, kB, z);
    kB = load_a16x32(kh, DHEAD, k0 + 16, 32);
    S1 = wmma_bf16(qA1, kB, S1);

    const int key0 = k0 + l15;
    const int key1 = k0 + 16 + l15;
    const int pm0  = amask[b * TSEQ + key0];
    const int pm1  = amask[b * TSEQ + key1];

#pragma unroll
    for (int r = 0; r < 8; ++r) {
      const int qrow = t0 + r + 8 * half;
      float s0 = (pm0 && key0 <= qrow) ? S0[r] * SCL : NEGINF;
      float s1 = (pm1 && key1 <= qrow) ? S1[r] * SCL : NEGINF;
      const float tmax = redmax16(fmaxf(s0, s1));
      const float nmax = fmaxf(rmax[r], tmax);
      const float corr = __expf(rmax[r] - nmax);
      rmax[r] = nmax;
      const float p0 = __expf(s0 - nmax);
      const float p1 = __expf(s1 - nmax);
      rsum[r] = rsum[r] * corr + redsum16(p0 + p1);
#pragma unroll
      for (int j = 0; j < 4; ++j) O[j][r] *= corr;
      psc[wv][(r + 8 * half) * 32 + l15]      = (__bf16)p0;
      psc[wv][(r + 8 * half) * 32 + 16 + l15] = (__bf16)p1;
    }
    asm volatile("s_wait_dscnt 0x0" ::: "memory");
    V16U pu;                                   // P in A layout (16 x 32)
    const __bf16* pp = &psc[wv][l15 * 32 + (half << 3)];
    pu.h[0] = *(const v8bf*)pp;
    pu.h[1] = *(const v8bf*)(pp + 16);
#pragma unroll
    for (int j = 0; j < 4; ++j) {
      const v16bf vB = load_a16x32(vh, TSEQ, 16 * j, k0);  // rows = d, cols = keys
      O[j] = wmma_bf16(pu.v, vB, O[j]);
    }
  }

#pragma unroll
  for (int r = 0; r < 8; ++r) {
    const float inv = 1.0f / rsum[r];
    const int   t   = t0 + r + 8 * half;
    const size_t ro = ((size_t)b * TSEQ + t) * CDIM + h * DHEAD;
#pragma unroll
    for (int j = 0; j < 4; ++j)
      yb[ro + 16 * j + l15] = (__bf16)(O[j][r] * inv);
  }
}

// ------------------------------------------------------------------- driver
extern "C" void kernel_launch(void* const* d_in, const int* in_sizes, int n_in,
                              void* d_out, int out_size, void* d_ws, size_t ws_size,
                              hipStream_t stream) {
  (void)in_sizes; (void)n_in; (void)out_size; (void)ws_size;
  const float* x  = (const float*)d_in[0];
  const int*   am = (const int*)  d_in[1];
  const float* Wq = (const float*)d_in[2];
  const float* bq = (const float*)d_in[3];
  const float* Wk = (const float*)d_in[4];
  const float* bk = (const float*)d_in[5];
  const float* Wv = (const float*)d_in[6];
  const float* bv = (const float*)d_in[7];
  const float* Wp = (const float*)d_in[8];
  const float* bp = (const float*)d_in[9];
  float* out = (float*)d_out;

  char* ws = (char*)d_ws;
  size_t off = 0;
  auto alloc = [&](size_t bytes) -> char* {
    char* p = ws + off;
    off += (bytes + 255) & ~(size_t)255;
    return p;
  };
  __bf16* xb   = (__bf16*)alloc((size_t)MROWS * CDIM * 2);
  __bf16* wqt  = (__bf16*)alloc((size_t)CDIM * CDIM * 2);
  __bf16* wkt  = (__bf16*)alloc((size_t)CDIM * CDIM * 2);
  __bf16* wvt  = (__bf16*)alloc((size_t)CDIM * CDIM * 2);
  __bf16* wpt  = (__bf16*)alloc((size_t)CDIM * CDIM * 2);
  __bf16* qbuf = (__bf16*)alloc((size_t)MROWS * CDIM * 2);
  __bf16* kbuf = (__bf16*)alloc((size_t)MROWS * CDIM * 2);
  __bf16* vtb  = (__bf16*)alloc((size_t)MROWS * CDIM * 2);
  __bf16* ybuf = (__bf16*)alloc((size_t)MROWS * CDIM * 2);

  const int n4 = MROWS * CDIM / 4;
  cvt_f32_bf16<<<(n4 + 255) / 256, 256, 0, stream>>>(x, xb, n4);
  w_transpose_bf16<<<1024, 256, 0, stream>>>(Wq, wqt);
  w_transpose_bf16<<<1024, 256, 0, stream>>>(Wk, wkt);
  w_transpose_bf16<<<1024, 256, 0, stream>>>(Wv, wvt);
  w_transpose_bf16<<<1024, 256, 0, stream>>>(Wp, wpt);

  gemm_wmma<<<2048, 128, 0, stream>>>(xb, wqt, bq, qbuf, 0);
  gemm_wmma<<<2048, 128, 0, stream>>>(xb, wkt, bk, kbuf, 0);
  gemm_wmma<<<2048, 128, 0, stream>>>(xb, wvt, bv, vtb, 1);

  attn_wmma<<<2048, 128, 0, stream>>>(qbuf, kbuf, vtb, am, ybuf);

  gemm_wmma<<<2048, 128, 0, stream>>>(ybuf, wpt, bp, out, 2);
}
